// NeuralPrecond_30039001268520
// MI455X (gfx1250) — matrix operand order repaired and checked by reading, not compiled, not run on hardware
//
#include <hip/hip_runtime.h>
#include <math.h>

// ---------------------------------------------------------------------------
// Problem constants (from reference)
// ---------------------------------------------------------------------------
#define NN 50000
#define NE 400000
#define NK 6

typedef __bf16 bf16;
typedef __attribute__((ext_vector_type(16))) __bf16 v16bf;
typedef __attribute__((ext_vector_type(8)))  float  v8f;

// ---------------------------------------------------------------------------
// Fragment loaders (CDNA5 WMMA 16x16x32 bf16 layouts, CDNA5 ISA)
// A 16x32: lane m=l&15; half=l>>4 -> kbase = half?8:0 ; elems 0..7 = K kbase..kbase+7,
//          elems 8..15 = K kbase+16..kbase+23  (two contiguous 16B chunks)
// B 32x16 (pre-swizzled to [kchunk][ntile][lane][16] contiguous per lane)
// C/D 16x16: lane n=l&15; VGPR j -> row (half? 8+j : j)
// ---------------------------------------------------------------------------
__device__ __forceinline__ v16bf load_afrag(const bf16* p) {
  union { v16bf v; uint4 q[2]; } u;
  u.q[0] = *(const uint4*)(p);
  u.q[1] = *(const uint4*)(p + 16);
  return u.v;
}
__device__ __forceinline__ v16bf load_bfrag(const bf16* p) {
  union { v16bf v; uint4 q[2]; } u;
  u.q[0] = *(const uint4*)(p);
  u.q[1] = *(const uint4*)(p + 8);
  return u.v;
}

template<int NT>
__device__ __forceinline__ void accum_src(v8f* acc, const bf16* A, const int* I, int kc,
                                          int& kg, int rowA, int kbase,
                                          const bf16* Wsw, int lane)
{
  if (A == nullptr) return;                      // uniform branch
  long rstride = (long)kc * 32;
  long r = I ? (long)I[rowA] : (long)rowA;
  const bf16* base = A + r * rstride;
  for (int c = 0; c < kc; ++c, ++kg) {
    v16bf a = load_afrag(base + c * 32 + kbase);
    const bf16* wp = Wsw + (long)kg * NT * 512 + lane * 16;
#pragma unroll
    for (int t = 0; t < NT; ++t) {
      v16bf b = load_bfrag(wp + t * 512);
      acc[t] = __builtin_amdgcn_wmma_f32_16x16x32_bf16(false, a, false, b,
                                                       (short)0, acc[t], false, false);
    }
  }
}

// Generic fused GEMM:  out = act( sum_s A_s[idx_s] @ W_s + bias ) (*gate)(+resid)
// optional dense f32/bf16 stores and atomic scatter-add (segment sum) epilogue.
// ACT: 0=none 1=relu 2=sigmoid.  Nout = NT*16.
template<int ACT, int NT>
__global__ __launch_bounds__(256) void gemm_bf16_wmma(
    const bf16* A0, const int* I0, int kc0,
    const bf16* A1, const int* I1, int kc1,
    const bf16* A2, const int* I2, int kc2,
    const bf16* Wsw, const float* bias, int M,
    float* outF, bf16* outBf,
    const float* gateMul, const float* resid,
    float* scat, const int* sidx)
{
  const int Nout = NT * 16;
  int lane = threadIdx.x & 31;
  int wv   = threadIdx.x >> 5;
  int r0   = (blockIdx.x * 8 + wv) << 4;
  if (r0 >= M) return;                           // wave-uniform
  int m    = lane & 15;
  int half = lane >> 4;
  int rowA = r0 + m; if (rowA >= M) rowA = M - 1;
  int kbase = half ? 8 : 0;

  v8f zero = {0.f,0.f,0.f,0.f,0.f,0.f,0.f,0.f};
  v8f acc[NT];
#pragma unroll
  for (int t = 0; t < NT; ++t) acc[t] = zero;

  int kg = 0;
  accum_src<NT>(acc, A0, I0, kc0, kg, rowA, kbase, Wsw, lane);
  accum_src<NT>(acc, A1, I1, kc1, kg, rowA, kbase, Wsw, lane);
  accum_src<NT>(acc, A2, I2, kc2, kg, rowA, kbase, Wsw, lane);

#pragma unroll
  for (int t = 0; t < NT; ++t) {
    int n = t * 16 + m;
    float bs = bias[n];
#pragma unroll
    for (int j = 0; j < 8; ++j) {
      int rr = r0 + half * 8 + j;
      if (rr < M) {
        float v = acc[t][j] + bs;
        if (ACT == 1) v = v > 0.f ? v : 0.f;
        if (ACT == 2) v = 1.f / (1.f + __expf(-v));
        if (gateMul) v *= gateMul[(long)rr * Nout + n];
        if (resid)   v += resid[(long)rr * Nout + n];
        if (outF)   outF[(long)rr * Nout + n] = v;
        if (outBf)  outBf[(long)rr * Nout + n] = (bf16)v;
        if (scat)   atomicAdd(&scat[(long)sidx[rr] * Nout + n], v);
      }
    }
  }
}

// ---------------------------------------------------------------------------
// Support kernels
// ---------------------------------------------------------------------------

// f32 row-major W (K x Nout) -> swizzled bf16 fragments [kchunk][ntile][lane][16]
__global__ void swizzle_w_kernel(const float* W, bf16* Wsw, int K, int Nout)
{
  int i = blockIdx.x * blockDim.x + threadIdx.x;
  if (i >= K * Nout) return;
  int ntiles = Nout >> 4;
  int e    = i & 15;
  int lane = (i >> 4) & 31;
  int t    = i >> 9;
  int nt   = t % ntiles;
  int kc   = t / ntiles;
  int n = nt * 16 + (lane & 15);
  int k = kc * 32 + ((lane >> 4) << 3) + (e & 7) + ((e >> 3) << 4);
  Wsw[i] = (bf16)W[k * Nout + n];
}

// small-K dense layer (relu) -> bf16 out, one thread per output element
__global__ void dense_small_kernel(const float* X, const float* W, const float* b,
                                   bf16* out, int M, int Kin, int Nh)
{
  int i = blockIdx.x * blockDim.x + threadIdx.x;
  if (i >= M * Nh) return;
  int r = i / Nh, j = i - r * Nh;
  float s = b[j];
  const float* xr = X + (long)r * Kin;
  for (int k = 0; k < Kin; ++k) s += xr[k] * W[k * Nh + j];
  out[i] = (bf16)(s > 0.f ? s : 0.f);
}

// apply_input layer0: concat([node_cache(32), v_re, v_im]) @ W(34x32) + b, relu -> bf16
__global__ void apply_in1_kernel(const float* nc, const float* vre, const float* vim,
                                 const float* W, const float* b, bf16* out, int N)
{
  int i = blockIdx.x * blockDim.x + threadIdx.x;
  if (i >= N * 32) return;
  int r = i >> 5, j = i & 31;
  float s = b[j];
  const float* xr = nc + (long)r * 32;
  for (int k = 0; k < 32; ++k) s += xr[k] * W[k * 32 + j];
  s += vre[r] * W[32 * 32 + j];
  s += vim[r] * W[33 * 32 + j];
  out[i] = (bf16)(s > 0.f ? s : 0.f);
}

__global__ void global_encode_kernel(const float* g, const float* W0, const float* b0,
                                     const float* W1, const float* b1, float* gl)
{
  __shared__ float hid[64];
  int j = threadIdx.x;
  float s = b0[j];
  for (int k = 0; k < 4; ++k) s += g[k] * W0[k * 64 + j];
  hid[j] = s > 0.f ? s : 0.f;
  __syncthreads();
  float o = b1[j];
  for (int k = 0; k < 64; ++k) o += hid[k] * W1[k * 64 + j];
  gl[j] = o;
}

__global__ void global_update_kernel(float* gl, const float* hsum, const float* esum,
                                     float invN, float invE,
                                     const float* W0, const float* b0,
                                     const float* W1, const float* b1)
{
  __shared__ float in[192];
  __shared__ float hid[64];
  int j = threadIdx.x;
  in[j]       = gl[j];
  in[64 + j]  = hsum[j] * invN;
  in[128 + j] = esum[j] * invE;
  __syncthreads();
  float s = b0[j];
  for (int k = 0; k < 192; ++k) s += in[k] * W0[k * 64 + j];
  hid[j] = s > 0.f ? s : 0.f;
  __syncthreads();
  float o = b1[j];
  for (int k = 0; k < 64; ++k) o += hid[k] * W1[k * 64 + j];
  gl[j] = o;
}

// bias' = b + gl @ W[koff:koff+64]  (folds broadcast global term into bias)
__global__ void fold_bias_kernel(const float* W, const float* b, const float* gl,
                                 float* out, int koff)
{
  int n = threadIdx.x;  // 64 threads
  float s = b[n];
  for (int k = 0; k < 64; ++k) s += gl[k] * W[(koff + k) * 64 + n];
  out[n] = s;
}

__global__ void zero_kernel(float* p, int n)
{
  int i = blockIdx.x * blockDim.x + threadIdx.x;
  int step = gridDim.x * blockDim.x;
  for (; i < n; i += step) p[i] = 0.f;
}

__global__ void count_kernel(const int* row, float* cnt, int E)
{
  int i = blockIdx.x * blockDim.x + threadIdx.x;
  if (i < E) atomicAdd(&cnt[row[i]], 1.f);
}

// agg / max(cnt,1) -> bf16
__global__ void seg_div_kernel(const float* agg, const float* cnt, bf16* out, int n, int C)
{
  int i = blockIdx.x * blockDim.x + threadIdx.x;
  if (i >= n * C) return;
  int r = i / C;
  float c = cnt[r]; if (c < 1.f) c = 1.f;
  out[i] = (bf16)(agg[i] / c);
}

// column sums of a M x 64 bf16 matrix into sums[64] (pre-zeroed)
__global__ void colsum_kernel(const bf16* A, int M, float* sums)
{
  int tid = blockIdx.x * blockDim.x + threadIdx.x;
  int col = tid & 63;
  int r = tid >> 6;
  int step = (gridDim.x * blockDim.x) >> 6;
  float acc = 0.f;
  for (; r < M; r += step) acc += (float)A[(long)r * 64 + col];
  atomicAdd(&sums[col], acc);
}

// corr = ha @ W(32x2) + b ; out = v + corr
__global__ void final_out_kernel(const float* ha, const float* W, const float* b,
                                 const float* vre, const float* vim, float* out, int N)
{
  int i = blockIdx.x * blockDim.x + threadIdx.x;
  if (i >= N) return;
  float c0 = b[0], c1 = b[1];
  const float* hr = ha + (long)i * 32;
  for (int k = 0; k < 32; ++k) { float h = hr[k]; c0 += h * W[k * 2]; c1 += h * W[k * 2 + 1]; }
  out[i]     = vre[i] + c0;
  out[N + i] = vim[i] + c1;
}

// ---------------------------------------------------------------------------
// Host side
// ---------------------------------------------------------------------------
struct Ptrs {
  const float *x, *edge_attr, *g, *v_re, *v_im;
  const int* edge_index;
  const float *ne_w0,*ne_b0,*ne_w1,*ne_b1;
  const float *ee_w0,*ee_b0,*ee_w1,*ee_b1;
  const float *ge_w0,*ge_b0,*ge_w1,*ge_b1;
  const float *pe_w0[NK],*pe_b0[NK],*pe_w1[NK],*pe_b1[NK];
  const float *pn_w0[NK],*pn_b0[NK],*pn_w1[NK],*pn_b1[NK];
  const float *pg_w0[NK],*pg_b0[NK],*pg_w1[NK],*pg_b1[NK];
  const float *sk_w[NK],*sk_b[NK];
  const float *proj_w,*proj_b,*gate_w,*gate_b;
  const float *ai_w0,*ai_b0,*ai_w1,*ai_b1;
  const float *msg_w,*msg_b,*upd_w0,*upd_b0,*upd_w1,*upd_b1;
  const float *ao_w,*ao_b;
};

static void launch_gemm(hipStream_t s, int act, int nt, int M,
                        const bf16* A0, const int* I0, int kc0,
                        const bf16* A1, const int* I1, int kc1,
                        const bf16* A2, const int* I2, int kc2,
                        const bf16* W, const float* bias,
                        float* outF, bf16* outBf,
                        const float* gate, const float* resid,
                        float* scat, const int* sidx)
{
  dim3 grid((M + 127) / 128), blk(256);
#define GEMM_GO(A_, T_) gemm_bf16_wmma<A_, T_><<<grid, blk, 0, s>>>( \
      A0, I0, kc0, A1, I1, kc1, A2, I2, kc2, W, bias, M, outF, outBf, gate, resid, scat, sidx)
  if (nt == 4) {
    if (act == 0) GEMM_GO(0, 4); else if (act == 1) GEMM_GO(1, 4); else GEMM_GO(2, 4);
  } else {
    if (act == 0) GEMM_GO(0, 2); else if (act == 1) GEMM_GO(1, 2); else GEMM_GO(2, 2);
  }
#undef GEMM_GO
}

static void launch_swz(hipStream_t s, const float* W, bf16* Wsw, int K, int Nout)
{
  int total = K * Nout;
  swizzle_w_kernel<<<(total + 255) / 256, 256, 0, s>>>(W, Wsw, K, Nout);
}

extern "C" void kernel_launch(void* const* d_in, const int* in_sizes, int n_in,
                              void* d_out, int out_size, void* d_ws, size_t ws_size,
                              hipStream_t stream)
{
  (void)out_size; (void)ws_size;
  const int N = NN, E = NE;
  Ptrs p{};
  auto F = [&](int i) { return (const float*)d_in[i]; };

  // params flattened with dicts in *sorted-key* order (jax tree_leaves style)
  auto setParamsSorted = [&](int i) {
    p.ai_b0=F(i++); p.ai_w0=F(i++); p.ai_b1=F(i++); p.ai_w1=F(i++);
    p.msg_b=F(i++); p.msg_w=F(i++); p.upd_b0=F(i++); p.upd_w0=F(i++); p.upd_b1=F(i++); p.upd_w1=F(i++);
    p.ao_b=F(i++); p.ao_w=F(i++);
    p.ee_b0=F(i++); p.ee_w0=F(i++); p.ee_b1=F(i++); p.ee_w1=F(i++);
    p.gate_b=F(i++); p.gate_w=F(i++);
    for (int k = 0; k < NK; ++k) { p.sk_b[k]=F(i++); p.sk_w[k]=F(i++); }
    p.ge_b0=F(i++); p.ge_w0=F(i++); p.ge_b1=F(i++); p.ge_w1=F(i++);
    p.ne_b0=F(i++); p.ne_w0=F(i++); p.ne_b1=F(i++); p.ne_w1=F(i++);
    p.proj_b=F(i++); p.proj_w=F(i++);
    for (int k = 0; k < NK; ++k) {
      p.pe_b0[k]=F(i++); p.pe_w0[k]=F(i++); p.pe_b1[k]=F(i++); p.pe_w1[k]=F(i++);
      p.pg_b0[k]=F(i++); p.pg_w0[k]=F(i++); p.pg_b1[k]=F(i++); p.pg_w1[k]=F(i++);
      p.pn_b0[k]=F(i++); p.pn_w0[k]=F(i++); p.pn_b1[k]=F(i++); p.pn_w1[k]=F(i++);
    }
    return i;
  };
  // params flattened in *insertion* order (dict order as written, lin = {w,b})
  auto setParamsInsertion = [&](int i) {
    p.ne_w0=F(i++); p.ne_b0=F(i++); p.ne_w1=F(i++); p.ne_b1=F(i++);
    p.ee_w0=F(i++); p.ee_b0=F(i++); p.ee_w1=F(i++); p.ee_b1=F(i++);
    p.ge_w0=F(i++); p.ge_b0=F(i++); p.ge_w1=F(i++); p.ge_b1=F(i++);
    for (int k = 0; k < NK; ++k) {
      p.pe_w0[k]=F(i++); p.pe_b0[k]=F(i++); p.pe_w1[k]=F(i++); p.pe_b1[k]=F(i++);
      p.pn_w0[k]=F(i++); p.pn_b0[k]=F(i++); p.pn_w1[k]=F(i++); p.pn_b1[k]=F(i++);
      p.pg_w0[k]=F(i++); p.pg_b0[k]=F(i++); p.pg_w1[k]=F(i++); p.pg_b1[k]=F(i++);
    }
    for (int k = 0; k < NK; ++k) { p.sk_w[k]=F(i++); p.sk_b[k]=F(i++); }
    p.proj_w=F(i++); p.proj_b=F(i++);
    p.gate_w=F(i++); p.gate_b=F(i++);
    p.ai_w0=F(i++); p.ai_b0=F(i++); p.ai_w1=F(i++); p.ai_b1=F(i++);
    p.msg_w=F(i++); p.msg_b=F(i++); p.upd_w0=F(i++); p.upd_b0=F(i++); p.upd_w1=F(i++); p.upd_b1=F(i++);
    p.ao_w=F(i++); p.ao_b=F(i++);
    return i;
  };

  if (n_in > 0 && in_sizes[0] == E * 8) {
    // fully-sorted tree_leaves: edge_attr, edge_index, g, <params>, v_im, v_re, x
    p.edge_attr = F(0);
    p.edge_index = (const int*)d_in[1];
    p.g = F(2);
    int i = setParamsSorted(3);
    p.v_im = F(i++); p.v_re = F(i++); p.x = F(i++);
  } else {
    // insertion-order top level: x, edge_attr, g, v_re, v_im, edge_index, <params>
    p.x = F(0); p.edge_attr = F(1); p.g = F(2);
    p.v_re = F(3); p.v_im = F(4);
    p.edge_index = (const int*)d_in[5];
    if (n_in > 6 && in_sizes[6] == 9 * 64) setParamsInsertion(6);
    else setParamsSorted(6);
  }

  const int* row = p.edge_index;
  const int* col = p.edge_index + E;

  // ---------------- workspace layout ----------------
  char* wsb = (char*)d_ws;
  size_t off = 0;
  auto alloc = [&](size_t bytes) -> char* {
    off = (off + 255) & ~(size_t)255;
    char* ptr = wsb + off; off += bytes; return ptr;
  };
  bf16* h_bf       = (bf16*)alloc((size_t)N * 64 * 2);
  bf16* e_bf       = (bf16*)alloc((size_t)E * 64 * 2);
  bf16* e_enc_bf   = (bf16*)alloc((size_t)E * 64 * 2);
  bf16* edge_in_bf = (bf16*)alloc((size_t)E * 64 * 2);   // reused as gate f32 (E x 32) after loop
  bf16* hid_e_bf   = (bf16*)alloc((size_t)E * 64 * 2);
  bf16* hid_n_bf   = (bf16*)alloc((size_t)N * 64 * 2);   // reused as hid_a_bf (N x 32)
  float* agg       = (float*)alloc((size_t)N * 64 * 4);  // reused as agg2 (N x 32)
  bf16* agg_bf     = (bf16*)alloc((size_t)N * 64 * 2);
  float* cnt       = (float*)alloc((size_t)N * 4);
  float* node_cache= (float*)alloc((size_t)N * 32 * 4);
  float* ha        = (float*)alloc((size_t)N * 32 * 4);
  bf16* ha_bf      = (bf16*)alloc((size_t)N * 32 * 2);
  float* scal      = (float*)alloc(512 * 4);
  float* gl     = scal;
  float* hsum   = scal + 64;
  float* esum   = scal + 128;
  float* bias_e = scal + 192;
  float* bias_n = scal + 256;
  float* gate   = (float*)edge_in_bf;                    // E x 32 f32, same bytes as E x 64 bf16
  bf16* hid_a_bf = hid_n_bf;

  bf16* wsw_ne1  = (bf16*)alloc(64 * 64 * 2);
  bf16* wsw_ee1  = (bf16*)alloc(64 * 64 * 2);
  bf16* wsw_sk[NK]; bf16* wsw_pe0[NK]; bf16* wsw_pe1[NK]; bf16* wsw_pn0[NK]; bf16* wsw_pn1[NK];
  for (int k = 0; k < NK; ++k) {
    wsw_sk[k]  = (bf16*)alloc(128 * 64 * 2);
    wsw_pe0[k] = (bf16*)alloc(192 * 64 * 2);
    wsw_pe1[k] = (bf16*)alloc(64 * 64 * 2);
    wsw_pn0[k] = (bf16*)alloc(128 * 64 * 2);
    wsw_pn1[k] = (bf16*)alloc(64 * 64 * 2);
  }
  bf16* wsw_proj = (bf16*)alloc(64 * 32 * 2);
  bf16* wsw_gate = (bf16*)alloc(64 * 32 * 2);
  bf16* wsw_ai1  = (bf16*)alloc(32 * 32 * 2);
  bf16* wsw_msg  = (bf16*)alloc(32 * 32 * 2);
  bf16* wsw_upd0 = (bf16*)alloc(64 * 32 * 2);
  bf16* wsw_upd1 = (bf16*)alloc(32 * 32 * 2);

  // ---------------- weight prep (swizzle f32 -> bf16 fragments) ----------------
  launch_swz(stream, p.ne_w1, wsw_ne1, 64, 64);
  launch_swz(stream, p.ee_w1, wsw_ee1, 64, 64);
  for (int k = 0; k < NK; ++k) {
    launch_swz(stream, p.sk_w[k],  wsw_sk[k], 128, 64);
    launch_swz(stream, p.pe_w0[k], wsw_pe0[k], 192, 64);  // first 3 K-blocks; gl block folded into bias
    launch_swz(stream, p.pe_w1[k], wsw_pe1[k], 64, 64);
    launch_swz(stream, p.pn_w0[k], wsw_pn0[k], 128, 64);  // [h, agg] blocks
    launch_swz(stream, p.pn_w1[k], wsw_pn1[k], 64, 64);
  }
  launch_swz(stream, p.proj_w, wsw_proj, 64, 32);
  launch_swz(stream, p.gate_w, wsw_gate, 64, 32);
  launch_swz(stream, p.ai_w1,  wsw_ai1, 32, 32);
  launch_swz(stream, p.msg_w,  wsw_msg, 32, 32);
  launch_swz(stream, p.upd_w0, wsw_upd0, 64, 32);
  launch_swz(stream, p.upd_w1, wsw_upd1, 32, 32);

  // ---------------- segment counts (row histogram, fixed across steps) --------
  zero_kernel<<<256, 256, 0, stream>>>(cnt, N);
  count_kernel<<<(E + 255) / 256, 256, 0, stream>>>(row, cnt, E);

  // ---------------- encoders ----------------
  global_encode_kernel<<<1, 64, 0, stream>>>(p.g, p.ge_w0, p.ge_b0, p.ge_w1, p.ge_b1, gl);

  dense_small_kernel<<<(N * 64 + 255) / 256, 256, 0, stream>>>(p.x, p.ne_w0, p.ne_b0, hid_n_bf, N, 9, 64);
  launch_gemm(stream, 0, 4, N, hid_n_bf, nullptr, 2, nullptr, nullptr, 0, nullptr, nullptr, 0,
              wsw_ne1, p.ne_b1, nullptr, h_bf, nullptr, nullptr, nullptr, nullptr);

  dense_small_kernel<<<(E * 64 + 255) / 256, 256, 0, stream>>>(p.edge_attr, p.ee_w0, p.ee_b0, hid_e_bf, E, 8, 64);
  launch_gemm(stream, 0, 4, E, hid_e_bf, nullptr, 2, nullptr, nullptr, 0, nullptr, nullptr, 0,
              wsw_ee1, p.ee_b1, nullptr, e_bf, nullptr, nullptr, nullptr, nullptr);
  hipMemcpyAsync(e_enc_bf, e_bf, (size_t)E * 64 * 2, hipMemcpyDeviceToDevice, stream);

  // ---------------- message passing steps ----------------
  for (int k = 0; k < NK; ++k) {
    // edge_input = e @ Wsk[:64] + e_enc @ Wsk[64:128] + b
    launch_gemm(stream, 0, 4, E, e_bf, nullptr, 2, e_enc_bf, nullptr, 2, nullptr, nullptr, 0,
                wsw_sk[k], p.sk_b[k], nullptr, edge_in_bf, nullptr, nullptr, nullptr, nullptr);
    // fold gl term of edge MLP layer0 into bias
    fold_bias_kernel<<<1, 64, 0, stream>>>(p.pe_w0[k], p.pe_b0[k], gl, bias_e, 192);
    // hidden = relu(edge_input@W0 + h[row]@W1 + h[col]@W2 + bias')
    launch_gemm(stream, 1, 4, E, edge_in_bf, nullptr, 2, h_bf, row, 2, h_bf, col, 2,
                wsw_pe0[k], bias_e, nullptr, hid_e_bf, nullptr, nullptr, nullptr, nullptr);
    // e = hidden @ W1 + b1 ; fused segment-sum scatter into agg[row]
    zero_kernel<<<256, 256, 0, stream>>>(agg, N * 64);
    launch_gemm(stream, 0, 4, E, hid_e_bf, nullptr, 2, nullptr, nullptr, 0, nullptr, nullptr, 0,
                wsw_pe1[k], p.pe_b1[k], nullptr, e_bf, nullptr, nullptr, agg, row);
    seg_div_kernel<<<(N * 64 + 255) / 256, 256, 0, stream>>>(agg, cnt, agg_bf, N, 64);
    // node update
    fold_bias_kernel<<<1, 64, 0, stream>>>(p.pn_w0[k], p.pn_b0[k], gl, bias_n, 128);
    launch_gemm(stream, 1, 4, N, h_bf, nullptr, 2, agg_bf, nullptr, 2, nullptr, nullptr, 0,
                wsw_pn0[k], bias_n, nullptr, hid_n_bf, nullptr, nullptr, nullptr, nullptr);
    launch_gemm(stream, 0, 4, N, hid_n_bf, nullptr, 2, nullptr, nullptr, 0, nullptr, nullptr, 0,
                wsw_pn1[k], p.pn_b1[k], nullptr, h_bf, nullptr, nullptr, nullptr, nullptr);
    // global update
    zero_kernel<<<1, 256, 0, stream>>>(hsum, 128);   // hsum + esum
    colsum_kernel<<<256, 256, 0, stream>>>(h_bf, N, hsum);
    colsum_kernel<<<256, 256, 0, stream>>>(e_bf, E, esum);
    global_update_kernel<<<1, 64, 0, stream>>>(gl, hsum, esum, 1.f / N, 1.f / E,
                                               p.pg_w0[k], p.pg_b0[k], p.pg_w1[k], p.pg_b1[k]);
  }

  // ---------------- readout + apply ----------------
  launch_gemm(stream, 0, 2, N, h_bf, nullptr, 2, nullptr, nullptr, 0, nullptr, nullptr, 0,
              wsw_proj, p.proj_b, node_cache, nullptr, nullptr, nullptr, nullptr, nullptr);
  launch_gemm(stream, 2, 2, E, e_bf, nullptr, 2, nullptr, nullptr, 0, nullptr, nullptr, 0,
              wsw_gate, p.gate_b, gate, nullptr, nullptr, nullptr, nullptr, nullptr);

  apply_in1_kernel<<<(N * 32 + 255) / 256, 256, 0, stream>>>(node_cache, p.v_re, p.v_im,
                                                             p.ai_w0, p.ai_b0, hid_a_bf, N);
  launch_gemm(stream, 0, 2, N, hid_a_bf, nullptr, 1, nullptr, nullptr, 0, nullptr, nullptr, 0,
              wsw_ai1, p.ai_b1, ha, ha_bf, nullptr, nullptr, nullptr, nullptr);

  // msg = (ha[col] @ Wmsg + b) * gate ; segment-sum into agg2[row]  (no dense store)
  zero_kernel<<<256, 256, 0, stream>>>(agg, N * 32);
  launch_gemm(stream, 0, 2, E, ha_bf, col, 1, nullptr, nullptr, 0, nullptr, nullptr, 0,
              wsw_msg, p.msg_b, nullptr, nullptr, gate, nullptr, agg, row);
  seg_div_kernel<<<(N * 32 + 255) / 256, 256, 0, stream>>>(agg, cnt, agg_bf, N, 32);

  // ha = ha + MLP([ha, agg])
  launch_gemm(stream, 1, 2, N, ha_bf, nullptr, 1, agg_bf, nullptr, 1, nullptr, nullptr, 0,
              wsw_upd0, p.upd_b0, nullptr, hid_a_bf, nullptr, nullptr, nullptr, nullptr);
  launch_gemm(stream, 0, 2, N, hid_a_bf, nullptr, 1, nullptr, nullptr, 0, nullptr, nullptr, 0,
              wsw_upd1, p.upd_b1, ha, nullptr, nullptr, ha, nullptr, nullptr);

  final_out_kernel<<<(N + 255) / 256, 256, 0, stream>>>(ha, p.ao_w, p.ao_b, p.v_re, p.v_im,
                                                        (float*)d_out, N);
}
